// PositionEncoder_9259949490698
// MI455X (gfx1250) — matrix operand order repaired and checked by reading, not compiled, not run on hardware
//
#include <hip/hip_runtime.h>

typedef __attribute__((ext_vector_type(16))) _Float16 v16h;
typedef __attribute__((ext_vector_type(8)))  _Float16 v8h;
typedef __attribute__((ext_vector_type(8)))  float    v8f;

namespace {

constexpr int kNQ = 5000, kNC = 20000, kKNN = 32, kE = kNQ * kKNN;

// ---- f16 weight layout inside d_ws (element offsets) ----
constexpr int GB_WV1 = 0, GB_WS = 4096, GB_WG = 86016, GB_WV2 = 102400, GB_SZ = 106496;
constexpr int W_ROOT = 0;
constexpr int W_NODE = W_ROOT + GB_SZ;
constexpr int W_EDGE = W_NODE + GB_SZ;      // Wv1 +0, Ws +4096 (64x128), Wg +12288, Wv2 +16384
constexpr int W_EDIR = W_EDGE + 20480;
constexpr int W_SCA  = W_EDIR + 4096;       // 256x64
constexpr int W_E2N  = W_SCA  + 16384;      // 64x64
constexpr int W_N2E  = W_E2N  + 4096;       // 64x256
constexpr int W_EVN  = W_N2E  + 16384;      // 64x64
constexpr int W_OUT  = W_EVN  + 4096;
constexpr int W_ATTQ = W_OUT  + GB_SZ;
constexpr int W_ATTK = W_ATTQ + GB_SZ;
constexpr int W_ATTV = W_ATTK + GB_SZ;
constexpr int W_TOTAL = W_ATTV + GB_SZ;     // 704512 halves

// ---- f32 intermediate layout (float offsets from ws base) ----
constexpr int F32_OFF = W_TOTAL / 2;        // 352256
constexpr int NS_OFF = F32_OFF;
constexpr int NV_OFF = NS_OFF + kNC * 256;
constexpr int QS_OFF = NV_OFF + kNC * 192;
constexpr int QV_OFF = QS_OFF + kNQ * 256;

} // namespace

struct GBW { const _Float16 *Wv1, *Ws, *Wg, *Wv2; const float* bg; };

// ---------------------------------------------------------------------------
// Block-wide GEMM: D[M x N] = A[M x K] * W^T (+bias), W is (N x K) f16 (L2-hot),
// A is f16 in LDS (row stride lda), D is f32 in LDS (row stride ldd).
// M multiple of 32, N multiple of 16, K multiple of 32.
// Each work item fuses TWO m-tiles: one B fragment load feeds two independent
// v_wmma accumulators (halves B traffic, fills the XDL pipe hazard-free).
// ---------------------------------------------------------------------------
template <bool HASB>
__device__ __forceinline__ void gemm_block(const _Float16* A, int lda,
                                           const _Float16* W,
                                           float* D, int ldd,
                                           int M, int N, int K,
                                           const float* bias) {
  const int lane = threadIdx.x & 31;
  const int wave = threadIdx.x >> 5;
  const int nw   = blockDim.x >> 5;
  const int nt = N >> 4;
  const int mg = M >> 5;                 // groups of 2 m-tiles
  const int ml = lane & 15;
  const int kh8  = (lane >> 4) << 3;     // A half-lane K offset (CDNA5 16-bit A layout)
  const int kh16 = (lane >> 4) << 3;     // B half-lane K offset (same table layout as A)
  for (int t = wave; t < mg * nt; t += nw) {
    const int m0 = (t / nt) << 5;
    const int n0 = (t % nt) << 4;
    v8f acc0 = {0.f, 0.f, 0.f, 0.f, 0.f, 0.f, 0.f, 0.f};
    v8f acc1 = acc0;
    const _Float16* pa0 = A + (m0 + ml) * lda + kh8;
    const _Float16* pa1 = pa0 + (lda << 4);
    const _Float16* pb  = W + (size_t)(n0 + ml) * (size_t)K + kh16;
    for (int k0 = 0; k0 < K; k0 += 32) {
      v8h blo  = *(const v8h*)(pb);
      v8h bhi  = *(const v8h*)(pb + 16);
      v8h a0lo = *(const v8h*)(pa0);
      v8h a0hi = *(const v8h*)(pa0 + 16);
      v8h a1lo = *(const v8h*)(pa1);
      v8h a1hi = *(const v8h*)(pa1 + 16);
      pb += 32; pa0 += 32; pa1 += 32;
      v16h a0, a1, b;
#pragma unroll
      for (int h = 0; h < 8; ++h) {
        b[h]  = blo[h];  b[h + 8]  = bhi[h];
        a0[h] = a0lo[h]; a0[h + 8] = a0hi[h];
        a1[h] = a1lo[h]; a1[h + 8] = a1hi[h];
      }
      acc0 = __builtin_amdgcn_wmma_f32_16x16x32_f16(false, a0, false, b, (short)0, acc0, false, false);
      acc1 = __builtin_amdgcn_wmma_f32_16x16x32_f16(false, a1, false, b, (short)0, acc1, false, false);
    }
    const int ncol = n0 + ml;
    const int mrow = m0 + ((lane >> 4) << 3);
    float bv = 0.f;
    if (HASB) bv = bias[ncol];
#pragma unroll
    for (int r = 0; r < 8; ++r) {
      float v0 = acc0[r], v1 = acc1[r];
      if (HASB) { v0 += bv; v1 += bv; }
      D[(mrow + r) * ldd + ncol] = v0;
      D[(mrow + 16 + r) * ldd + ncol] = v1;
    }
  }
}

// ---------------------------------------------------------------------------
// GBLinear (gated bilinear) on a 32-row tile, dh = in_v = out_v = 64.
// in_v/out_v layout: [n][c*3+i]. astage: 32*320 f16; vtmp: 96*64 f32; t64: 32*64 f32.
// out_s/out_v may alias in_s/in_v (read-before-write is sync-separated).
// ---------------------------------------------------------------------------
__device__ void gb_tile(const GBW w, int ins, int outs,
                        const float* in_s, const float* in_v,
                        float* out_s, float* out_v,
                        _Float16* astage, float* vtmp, float* t64) {
  const int tid = threadIdx.x, nthr = blockDim.x;
  // (1) v_hid = vec x Wv1^T : flatten (n,i) -> 96 rows
  for (int idx = tid; idx < 96 * 64; idx += nthr) {
    int r = idx >> 6, c = idx & 63;
    astage[idx] = (_Float16)in_v[(r / 3) * 192 + c * 3 + (r % 3)];
  }
  __syncthreads();
  gemm_block<false>(astage, 64, w.Wv1, vtmp, 64, 96, 64, 64, nullptr);
  __syncthreads();
  // (2) v_norm
  for (int idx = tid; idx < 32 * 64; idx += nthr) {
    int n = idx >> 6, h = idx & 63;
    float s = 1e-12f;
#pragma unroll
    for (int i = 0; i < 3; ++i) { float v = vtmp[(n * 3 + i) * 64 + h]; s += v * v; }
    t64[idx] = sqrtf(s);
  }
  __syncthreads();
  // (3) out_s = [v_norm | sca] x Ws^T
  const int K2 = 64 + ins;
  for (int idx = tid; idx < 32 * K2; idx += nthr) {
    int n = idx / K2, k = idx - n * K2;
    float v = (k < 64) ? t64[(n << 6) + k] : in_s[n * ins + (k - 64)];
    astage[idx] = (_Float16)v;
  }
  __syncthreads();
  gemm_block<false>(astage, K2, w.Ws, out_s, outs, 32, outs, K2, nullptr);
  __syncthreads();
  // (4) gate = sigmoid(out_s x Wg^T + bg)
  for (int idx = tid; idx < 32 * outs; idx += nthr) astage[idx] = (_Float16)out_s[idx];
  __syncthreads();
  gemm_block<true>(astage, outs, w.Wg, t64, 64, 32, 64, outs, w.bg);
  __syncthreads();
  for (int idx = tid; idx < 32 * 64; idx += nthr) t64[idx] = 1.f / (1.f + __expf(-t64[idx]));
  // (5) out_v = gate * (v_hid x Wv2^T)
  for (int idx = tid; idx < 96 * 64; idx += nthr) astage[idx] = (_Float16)vtmp[idx];
  __syncthreads();
  gemm_block<false>(astage, 64, w.Wv2, vtmp, 64, 96, 64, 64, nullptr);
  __syncthreads();
  for (int idx = tid; idx < 32 * 192; idx += nthr) {
    int n = idx / 192, rem = idx - n * 192;
    int c = rem / 3, i = rem - c * 3;
    out_v[idx] = t64[(n << 6) + c] * vtmp[(n * 3 + i) * 64 + c];
  }
  __syncthreads();
}

// ---------------------------------------------------------------------------
__global__ void pe_cvt_kernel(const float* src, _Float16* dst, int n) {
  int i = blockIdx.x * 256 + threadIdx.x;
  if (i < n) dst[i] = (_Float16)src[i];
}

// mm_node over 32 nodes/block
__global__ void __launch_bounds__(256) pe_node_kernel(const float* sca_g, const float* vec_g,
                                                      GBW w, float* ns_g, float* nv_g) {
  __shared__ float sIn[14336];
  __shared__ float sOut[14336];
  __shared__ __align__(16) _Float16 sA[10240];
  __shared__ float sV[6144];
  __shared__ float sT[2048];
  const int n0 = blockIdx.x * 32, tid = threadIdx.x;
  float* inS = sIn;  float* inV = sIn + 8192;
  float* oS  = sOut; float* oV  = sOut + 8192;
  for (int idx = tid; idx < 8192; idx += 256) inS[idx] = sca_g[(size_t)n0 * 256 + idx];
  for (int idx = tid; idx < 6144; idx += 256) inV[idx] = vec_g[(size_t)n0 * 192 + idx];
  __syncthreads();
  gb_tile(w, 256, 256, inS, inV, oS, oV, sA, sV, sT);
  for (int idx = tid; idx < 8192; idx += 256) ns_g[(size_t)n0 * 256 + idx] = oS[idx];
  for (int idx = tid; idx < 6144; idx += 256) nv_g[(size_t)n0 * 192 + idx] = oV[idx];
}

// root expansion + root_lin + att_q over 32 queries/block (last block partial)
__global__ void __launch_bounds__(256) pe_query_kernel(const float* pos_q, const float* emb,
                                                       const float* wrootexp, GBW root, GBW attq,
                                                       float* qs_g, float* qv_g) {
  __shared__ float sIn[14336];
  __shared__ float sMid[14336];
  __shared__ __align__(16) _Float16 sA[10240];
  __shared__ float sV[6144];
  __shared__ float sT[2048];
  const int n0 = blockIdx.x * 32, tid = threadIdx.x;
  const int valid = (kNQ - n0 < 32) ? (kNQ - n0) : 32;
  float* inS = sIn; float* inV = sIn + 8192;
  for (int idx = tid; idx < 8192; idx += 256) {
    int m = idx >> 8;
    inS[idx] = (m < valid) ? emb[(size_t)n0 * 256 + idx] : 0.f;
  }
  for (int idx = tid; idx < 6144; idx += 256) {
    int m = idx / 192, rem = idx - m * 192;
    int c = rem / 3, i = rem - c * 3;
    float v = 0.f;
    if (m < valid) {
      const float* p = pos_q + (size_t)(n0 + m) * 3;
      float nrm = sqrtf(p[0] * p[0] + p[1] * p[1] + p[2] * p[2] + 1e-12f) + 1e-7f;
      v = p[i] / nrm * wrootexp[c];
    }
    inV[idx] = v;
  }
  __syncthreads();
  float* mS = sMid; float* mV = sMid + 8192;
  gb_tile(root, 256, 256, inS, inV, mS, mV, sA, sV, sT);
  gb_tile(attq, 256, 256, mS, mV, inS, inV, sA, sV, sT);
  for (int idx = tid; idx < 8192; idx += 256) { int m = idx >> 8; if (m < valid) qs_g[(size_t)n0 * 256 + idx] = inS[idx]; }
  for (int idx = tid; idx < 6144; idx += 256) { int m = idx / 192; if (m < valid) qv_g[(size_t)n0 * 192 + idx] = inV[idx]; }
}

// One block per query: full per-edge chain + fused segment softmax + scatter-sum.
__global__ void __launch_bounds__(256) pe_edge_kernel(
    const int* eidx, const float* pos_q, const float* cpx, const float* wvexp,
    GBW edgeGB, const _Float16* Wdir,
    const _Float16* WscaM, const float* scaB,
    const _Float16* We2n, const float* e2nB,
    const _Float16* Wn2e, const float* n2eB,
    const _Float16* Wevn,
    GBW outGB, GBW attK, GBW attV,
    const float* ns_g, const float* nv_g, const float* qs_g, const float* qv_g,
    float* out) {
  __shared__ float R1[14336];   // ns|nv  -> ms|mv
  __shared__ float RY[14336];   // ys|yv  -> ks|kv -> vs|vv
  __shared__ float R2[8192];    // esca|evec -> es|ev
  __shared__ __align__(16) _Float16 sA[10240];
  __shared__ float sV[6144];
  __shared__ float sT[2048];
  __shared__ float sT2[2048];
  __shared__ float distS[32]; __shared__ int jS[32];
  __shared__ float dirS[96];
  __shared__ float qbuf[448];
  __shared__ float aS[128], aV[128], wSm[128], wVm[128];

  const int q = blockIdx.x, tid = threadIdx.x;
  float* ns = R1;   float* nv = R1 + 8192;
  float* esca = R2; float* evec = R2 + 2048;
  float* ys = RY;   float* yv = RY + 8192;

  // ---- Ph0: geometry (i == q since edges are KNN-contiguous) ----
  if (tid < 32) {
    int j = eidx[kE + q * kKNN + tid];
    jS[tid] = j;
    float dx = pos_q[q * 3 + 0] - cpx[(size_t)j * 3 + 0];
    float dy = pos_q[q * 3 + 1] - cpx[(size_t)j * 3 + 1];
    float dz = pos_q[q * 3 + 2] - cpx[(size_t)j * 3 + 2];
    dirS[tid * 3 + 0] = dx; dirS[tid * 3 + 1] = dy; dirS[tid * 3 + 2] = dz;
    distS[tid] = sqrtf(dx * dx + dy * dy + dz * dz + 1e-12f);
  }
  __syncthreads();
  for (int idx = tid; idx < 2048; idx += 256) {       // Gaussian smearing + edge expansion
    int t = idx >> 6, c = idx & 63;
    float d = distS[t];
    float u = d - (float)c * (10.f / 63.f);
    esca[idx] = __expf(-19.845f * u * u);             // -0.5/(10/63)^2
    float s = wvexp[c] / (d + 1e-7f);
    evec[t * 192 + c * 3 + 0] = dirS[t * 3 + 0] * s;
    evec[t * 192 + c * 3 + 1] = dirS[t * 3 + 1] * s;
    evec[t * 192 + c * 3 + 2] = dirS[t * 3 + 2] * s;
  }
  // ---- gather node features to LDS ----
  for (int idx = tid; idx < 8192; idx += 256) ns[idx] = ns_g[(size_t)jS[idx >> 8] * 256 + (idx & 255)];
  for (int idx = tid; idx < 6144; idx += 256) { int t = idx / 192; nv[idx] = nv_g[(size_t)jS[t] * 192 + (idx - t * 192)]; }
  __syncthreads();
  // ---- Ph1: mm_edge GBLinear (in-place) ----
  gb_tile(edgeGB, 64, 64, esca, evec, esca, evec, sA, sV, sT);
  float* es = esca; float* ev = evec;
  // ---- Ph2: VN-LeakyReLU on ev, leaky_relu on es ----
  for (int idx = tid; idx < 6144; idx += 256) {
    int r = idx >> 6, c = idx & 63;
    sA[idx] = (_Float16)ev[(r / 3) * 192 + c * 3 + (r % 3)];
  }
  __syncthreads();
  gemm_block<false>(sA, 64, Wdir, sV, 64, 96, 64, 64, nullptr);
  __syncthreads();
  for (int idx = tid; idx < 2048; idx += 256) {
    int n = idx >> 6, o = idx & 63;
    float x0 = ev[n * 192 + o * 3 + 0], x1 = ev[n * 192 + o * 3 + 1], x2 = ev[n * 192 + o * 3 + 2];
    float d0 = sV[(n * 3 + 0) * 64 + o], d1 = sV[(n * 3 + 1) * 64 + o], d2 = sV[(n * 3 + 2) * 64 + o];
    float dot = x0 * d0 + x1 * d1 + x2 * d2;
    float f = (dot >= 0.f) ? 0.f : dot / (d0 * d0 + d1 * d1 + d2 * d2 + 1e-6f);
    ev[n * 192 + o * 3 + 0] = 0.2f * x0 + 0.8f * (x0 - f * d0);
    ev[n * 192 + o * 3 + 1] = 0.2f * x1 + 0.8f * (x1 - f * d1);
    ev[n * 192 + o * 3 + 2] = 0.2f * x2 + 0.8f * (x2 - f * d2);
    float s = es[idx];
    es[idx] = (s >= 0.f) ? s : 0.01f * s;
  }
  __syncthreads();
  // ---- Ph4: y_s = n_s * (e_s x Wsca^T + b) ----
  for (int idx = tid; idx < 2048; idx += 256) sA[idx] = (_Float16)es[idx];
  __syncthreads();
  gemm_block<true>(sA, 64, WscaM, ys, 256, 32, 256, 64, scaB);
  __syncthreads();
  for (int idx = tid; idx < 8192; idx += 256) ys[idx] *= ns[idx];
  // ---- Ph5: gates + e_v2 + y_v ----
  gemm_block<true>(sA, 64, We2n, sT, 64, 32, 64, 64, e2nB);      // g_e2n (sA still holds e_s)
  __syncthreads();
  for (int idx = tid; idx < 8192; idx += 256) sA[idx] = (_Float16)ns[idx];
  __syncthreads();
  gemm_block<true>(sA, 256, Wn2e, sT2, 64, 32, 64, 256, n2eB);   // g_n2e
  __syncthreads();
  for (int idx = tid; idx < 6144; idx += 256) {
    int r = idx >> 6, c = idx & 63;
    sA[idx] = (_Float16)ev[(r / 3) * 192 + c * 3 + (r % 3)];
  }
  __syncthreads();
  gemm_block<false>(sA, 64, Wevn, sV, 64, 96, 64, 64, nullptr);  // e_v2
  __syncthreads();
  for (int idx = tid; idx < 6144; idx += 256) {
    int n = idx / 192, rem = idx - n * 192;
    int o = rem / 3, i = rem - o * 3;
    yv[idx] = sT[(n << 6) + o] * nv[idx] + sT2[(n << 6) + o] * sV[(n * 3 + i) * 64 + o];
  }
  __syncthreads();
  // ---- Ph6: mm_out (dest reuses ns/nv region) ----
  float* ms = R1; float* mv = R1 + 8192;
  gb_tile(outGB, 256, 256, ys, yv, ms, mv, sA, sV, sT);
  // ---- Ph7: cosine cutoff ----
  for (int idx = tid; idx < 8192; idx += 256) {
    float d = distS[idx >> 8];
    float C = (d <= 10.f) ? 0.5f * (__cosf(d * 0.314159265f) + 1.f) : 0.f;
    ms[idx] *= C;
  }
  for (int idx = tid; idx < 6144; idx += 256) {
    float d = distS[idx / 192];
    float C = (d <= 10.f) ? 0.5f * (__cosf(d * 0.314159265f) + 1.f) : 0.f;
    mv[idx] *= C;
  }
  __syncthreads();
  // ---- Ph8a: att_k -> ks/kv (reuse ys/yv region) ----
  float* ks = RY; float* kv = RY + 8192;
  gb_tile(attK, 256, 256, ms, mv, ks, kv, sA, sV, sT);
  // ---- Ph9a: attention logits vs this query's q ----
  for (int idx = tid; idx < 448; idx += 256)
    qbuf[idx] = (idx < 256) ? qs_g[(size_t)q * 256 + idx] : qv_g[(size_t)q * 192 + idx - 256];
  __syncthreads();
  if (tid < 128) {
    int t = tid >> 2, h = tid & 3;
    float as = 0.f;
    for (int c = 0; c < 64; ++c) as += qbuf[h * 64 + c] * ks[t * 256 + h * 64 + c];
    aS[t * 4 + h] = as * 0.125f;                       // 1/sqrt(64)
    float av = 0.f;
    for (int r = 0; r < 48; ++r) av += qbuf[256 + h * 48 + r] * kv[t * 192 + h * 48 + r];
    aV[t * 4 + h] = av * 0.14433757f;                  // 1/sqrt(48)
  }
  __syncthreads();
  // ---- Ph8b: att_v -> vs/vv (overwrite ks/kv) ----
  float* vs = RY; float* vv = RY + 8192;
  gb_tile(attV, 256, 256, ms, mv, vs, vv, sA, sV, sT);
  // ---- Ph10: per-head softmax over the 32 contiguous edges (wave32 shuffles) ----
  if (tid < 32) {
    for (int h = 0; h < 4; ++h) {
      float a = aS[tid * 4 + h], m = a;
      for (int o = 16; o > 0; o >>= 1) m = fmaxf(m, __shfl_xor(m, o, 32));
      float e = __expf(a - m), s = e;
      for (int o = 16; o > 0; o >>= 1) s += __shfl_xor(s, o, 32);
      wSm[tid * 4 + h] = e / (s + 1e-6f);
      float a2 = aV[tid * 4 + h], m2 = a2;
      for (int o = 16; o > 0; o >>= 1) m2 = fmaxf(m2, __shfl_xor(m2, o, 32));
      float e2 = __expf(a2 - m2), s2 = e2;
      for (int o = 16; o > 0; o >>= 1) s2 += __shfl_xor(s2, o, 32);
      wVm[tid * 4 + h] = e2 / (s2 + 1e-6f);
    }
  }
  __syncthreads();
  // ---- Ph11: weighted scatter-sum -> output ----
  {
    int c = tid, h = c >> 6;
    float acc = 0.f;
    for (int t = 0; t < 32; ++t) acc += wSm[t * 4 + h] * vs[t * 256 + c];
    out[(size_t)q * 256 + c] = acc;
  }
  if (tid < 192) {
    int h = (tid / 3) >> 4;
    float acc = 0.f;
    for (int t = 0; t < 32; ++t) acc += wVm[t * 4 + h] * vv[t * 192 + tid];
    out[(size_t)kNQ * 256 + (size_t)q * 192 + tid] = acc;
  }
}

// ---------------------------------------------------------------------------
extern "C" void kernel_launch(void* const* d_in, const int* in_sizes, int n_in,
                              void* d_out, int out_size, void* d_ws, size_t ws_size,
                              hipStream_t stream) {
  (void)in_sizes; (void)n_in; (void)out_size; (void)ws_size;
  _Float16* wb = (_Float16*)d_ws;
  float* fb = (float*)d_ws;

  // Weight f32 -> f16 conversions (params flattened in setup_inputs insertion order).
  struct CE { int src, off, cnt; };
  static const CE tbl[] = {
    {8,  W_ROOT + GB_WV1, 4096}, {9,  W_ROOT + GB_WS, 81920}, {10, W_ROOT + GB_WG, 16384}, {12, W_ROOT + GB_WV2, 4096},
    {13, W_NODE + GB_WV1, 4096}, {14, W_NODE + GB_WS, 81920}, {15, W_NODE + GB_WG, 16384}, {17, W_NODE + GB_WV2, 4096},
    {18, W_EDGE + 0, 4096}, {19, W_EDGE + 4096, 8192}, {20, W_EDGE + 12288, 4096}, {22, W_EDGE + 16384, 4096},
    {23, W_EDIR, 4096}, {24, W_SCA, 16384}, {26, W_E2N, 4096}, {28, W_N2E, 16384}, {30, W_EVN, 4096},
    {31, W_OUT  + GB_WV1, 4096}, {32, W_OUT  + GB_WS, 81920}, {33, W_OUT  + GB_WG, 16384}, {35, W_OUT  + GB_WV2, 4096},
    {36, W_ATTQ + GB_WV1, 4096}, {37, W_ATTQ + GB_WS, 81920}, {38, W_ATTQ + GB_WG, 16384}, {40, W_ATTQ + GB_WV2, 4096},
    {41, W_ATTK + GB_WV1, 4096}, {42, W_ATTK + GB_WS, 81920}, {43, W_ATTK + GB_WG, 16384}, {45, W_ATTK + GB_WV2, 4096},
    {46, W_ATTV + GB_WV1, 4096}, {47, W_ATTV + GB_WS, 81920}, {48, W_ATTV + GB_WG, 16384}, {50, W_ATTV + GB_WV2, 4096},
  };
  for (const CE& t : tbl)
    pe_cvt_kernel<<<(t.cnt + 255) / 256, 256, 0, stream>>>((const float*)d_in[t.src], wb + t.off, t.cnt);

  auto mkGB = [&](int base, int bgidx) {
    GBW g; g.Wv1 = wb + base + GB_WV1; g.Ws = wb + base + GB_WS;
    g.Wg = wb + base + GB_WG; g.Wv2 = wb + base + GB_WV2;
    g.bg = (const float*)d_in[bgidx]; return g;
  };
  GBW root = mkGB(W_ROOT, 11), node = mkGB(W_NODE, 16);
  GBW outg = mkGB(W_OUT, 34), attq = mkGB(W_ATTQ, 39), attk = mkGB(W_ATTK, 44), attv = mkGB(W_ATTV, 49);
  GBW edge; edge.Wv1 = wb + W_EDGE; edge.Ws = wb + W_EDGE + 4096;
  edge.Wg = wb + W_EDGE + 12288; edge.Wv2 = wb + W_EDGE + 16384;
  edge.bg = (const float*)d_in[21];

  pe_node_kernel<<<kNC / 32, 256, 0, stream>>>((const float*)d_in[3], (const float*)d_in[4], node,
                                               fb + NS_OFF, fb + NV_OFF);
  pe_query_kernel<<<(kNQ + 31) / 32, 256, 0, stream>>>((const float*)d_in[0], (const float*)d_in[5],
                                                       (const float*)d_in[7], root, attq,
                                                       fb + QS_OFF, fb + QV_OFF);
  pe_edge_kernel<<<kNQ, 256, 0, stream>>>((const int*)d_in[1], (const float*)d_in[0], (const float*)d_in[2],
                                          (const float*)d_in[6],
                                          edge, wb + W_EDIR,
                                          wb + W_SCA, (const float*)d_in[25],
                                          wb + W_E2N, (const float*)d_in[27],
                                          wb + W_N2E, (const float*)d_in[29],
                                          wb + W_EVN,
                                          outg, attk, attv,
                                          fb + NS_OFF, fb + NV_OFF, fb + QS_OFF, fb + QV_OFF,
                                          (float*)d_out);
}
